// MultiHeadAttention_81905026335339
// MI455X (gfx1250) — matrix operand order repaired and checked by reading, not compiled, not run on hardware
//
#include <hip/hip_runtime.h>
#include <math.h>

typedef __attribute__((ext_vector_type(16))) _Float16 v16h;
typedef __attribute__((ext_vector_type(8)))  _Float16 v8h;
typedef __attribute__((ext_vector_type(2)))  _Float16 v2h;
typedef __attribute__((ext_vector_type(8)))  float    v8f;

#define B_   2
#define S_   2048
#define D_   1024
#define H_   16
#define HD_  64
#define M_   (B_ * S_)   // 4096 rows
#define N3D_ (3 * D_)    // 3072

union AFrag { v16h v; v8h h[2]; };

static __device__ __forceinline__ v8f wmma_f16(v16h a, v16h b, v8f c) {
  // D = A(16x32 f16) * B(32x16 f16) + C(16x16 f32)
  return __builtin_amdgcn_wmma_f32_16x16x32_f16(false, a, false, b, (short)0, c, false, false);
}

// ---------------------------------------------------------------- utilities
__global__ void cvt_f16_kernel(const float* __restrict__ src,
                               _Float16* __restrict__ dst, int n) {
  int i = blockIdx.x * blockDim.x + threadIdx.x;
  if (i < n) dst[i] = (_Float16)src[i];
}

// src is K x N (row-major f32) -> dst is N x K (row-major f16)
__global__ void transpose_f16_kernel(const float* __restrict__ src,
                                     _Float16* __restrict__ dst, int K, int N) {
  int i = blockIdx.x * blockDim.x + threadIdx.x;
  if (i < N * K) {
    int n = i / K, k = i - n * K;
    dst[i] = (_Float16)src[(size_t)k * N + n];
  }
}

__global__ void log_kernel(const float* __restrict__ mask,
                           float* __restrict__ lm, int n) {
  int i = blockIdx.x * blockDim.x + threadIdx.x;
  if (i < n) lm[i] = __logf(mask[i]);
}

// ---------------------------------------------------------------- WMMA GEMM
// C[M x N] = A[M x K](f16 row-major) * Bt[N x K]^T (f16, N-major) + bias[N]
// One wave computes a 32(M) x 64(N) tile: 8 accumulators, 8 WMMAs / k-step.
template <bool OUT_HALF>
__global__ __launch_bounds__(256) void gemm_kernel(
    const _Float16* __restrict__ A, const _Float16* __restrict__ Bt,
    const float* __restrict__ bias, void* __restrict__ out,
    int M, int N, int K) {
  const int lane = threadIdx.x & 31;
  const int wid  = blockIdx.x * (blockDim.x >> 5) + (threadIdx.x >> 5);
  const int tilesN = N >> 6;
  const int tilesM = M >> 5;
  if (wid >= tilesM * tilesN) return;   // wave-uniform: EXEC stays all-ones
  const int tm = wid / tilesN, tn = wid - tm * tilesN;
  const int mBase = tm << 5, nBase = tn << 6;
  const int g = lane >> 4, nl = lane & 15;
  const int kb = g * 8, kc = g * 16;

  v8f acc[2][4] = {};
  for (int k0 = 0; k0 < K; k0 += 32) {
    if (k0 + 64 <= K) {  // hide L2 latency one k-step ahead
      __builtin_prefetch(A + (size_t)(mBase + nl) * K + k0 + 32 + kb, 0, 0);
      __builtin_prefetch(Bt + (size_t)(nBase + nl) * K + k0 + 32 + kc, 0, 0);
    }
    AFrag a[2];
#pragma unroll
    for (int mt = 0; mt < 2; ++mt) {
      const _Float16* pa = A + (size_t)(mBase + mt * 16 + nl) * K + k0 + kc;
      a[mt].h[0] = *(const v8h*)(pa);
      a[mt].h[1] = *(const v8h*)(pa + 8);
    }
#pragma unroll
    for (int nt = 0; nt < 4; ++nt) {
      const _Float16* pb = Bt + (size_t)(nBase + nt * 16 + nl) * K + k0 + kc;
      v16h bf = *(const v16h*)pb;
      acc[0][nt] = wmma_f16(a[0].v, bf, acc[0][nt]);
      acc[1][nt] = wmma_f16(a[1].v, bf, acc[1][nt]);
    }
  }
#pragma unroll
  for (int nt = 0; nt < 4; ++nt) {
    const int n = nBase + nt * 16 + nl;
    const float bv = bias[n];
#pragma unroll
    for (int mt = 0; mt < 2; ++mt) {
#pragma unroll
      for (int r = 0; r < 8; ++r) {
        const int m = mBase + mt * 16 + r + 8 * g;
        const float v = acc[mt][nt][r] + bv;
        if (OUT_HALF) ((_Float16*)out)[(size_t)m * N + n] = (_Float16)v;
        else          ((float*)out)[(size_t)m * N + n]    = v;
      }
    }
  }
}

// ---------------------------------------------------------------- attention
// qkv: (B,S,3,H,HD) f16 row stride 3072. One wave per (b,h,16-row block),
// online (flash) softmax over 32-key steps, causal.
__global__ __launch_bounds__(256) void attn_kernel(
    const _Float16* __restrict__ qkv, const float* __restrict__ logm,
    _Float16* __restrict__ attnh) {
  __shared__ _Float16 lds[8 * 2560];          // per wave: 512 (P) + 2048 (V^T)
  const int lane = threadIdx.x & 31;
  const int widx = threadIdx.x >> 5;
  _Float16* ldsP = &lds[widx * 2560];
  _Float16* ldsV = ldsP + 512;

  const int waveId   = blockIdx.x * 8 + widx;
  const int rowBlock = waveId & 127;          // S/16 = 128
  const int hb       = waveId >> 7;
  const int b        = hb >> 4;               // H = 16
  const int h        = hb & 15;
  const int rowBase  = rowBlock << 4;
  const int g = lane >> 4, nl = lane & 15;

  const _Float16* base = qkv + (size_t)b * S_ * 3072;
  const float*    lmb  = logm + b * S_;

  // Q fragments (16 rows x 64 features = two 16x32 A operands), contiguous k = 16*g + s order
  AFrag aq[2];
  {
    const _Float16* qrow = base + (size_t)(rowBase + nl) * 3072 + h * HD_;
    const int kb = g * 16;
#pragma unroll
    for (int c = 0; c < 2; ++c) {
      aq[c].h[0] = *(const v8h*)(qrow + c * 32 + kb);
      aq[c].h[1] = *(const v8h*)(qrow + c * 32 + kb + 8);
    }
  }
  float lmq[8];
#pragma unroll
  for (int r = 0; r < 8; ++r) lmq[r] = lmb[rowBase + r + 8 * g];

  float rmax[8], rsum[8];
#pragma unroll
  for (int r = 0; r < 8; ++r) { rmax[r] = -INFINITY; rsum[r] = 0.f; }
  v8f o[4] = {};

  const int nSteps = (rowBase + 16 + 31) >> 5;
  for (int st = 0; st < nSteps; ++st) {
    const int zBase = st << 5;

    // Stage V tile (32 keys x 64 dims) transposed into LDS: ldsV[d*32 + z].
    // Each lane owns two adjacent key rows and packs z-pairs -> ds_store_b32
    // (full 32-bit LDS bank utilization, half the DS ops of b16 scatter).
    {
      const int zp = (lane & 15) * 2;          // even key row pair
      const int dh = (lane >> 4) * 32;         // which half of the 64 dims
      const _Float16* v0 =
          base + (size_t)(zBase + zp) * 3072 + 2 * D_ + h * HD_ + dh;
      const _Float16* v1 = v0 + 3072;
      v8h r0[4], r1[4];
#pragma unroll
      for (int c = 0; c < 4; ++c) {
        r0[c] = *(const v8h*)(v0 + c * 8);
        r1[c] = *(const v8h*)(v1 + c * 8);
      }
#pragma unroll
      for (int j = 0; j < 32; ++j) {
        v2h pk;
        pk.x = r0[j >> 3][j & 7];
        pk.y = r1[j >> 3][j & 7];
        *(v2h*)(ldsV + (dh + j) * 32 + zp) = pk;
      }
    }

    // Prefetch next step's K and V rows into cache while we compute.
    if (st + 1 < nSteps) {
      __builtin_prefetch(
          base + (size_t)(zBase + 32 + lane) * 3072 + D_ + h * HD_, 0, 0);
      __builtin_prefetch(
          base + (size_t)(zBase + 32 + lane) * 3072 + 2 * D_ + h * HD_, 0, 0);
    }

    // Scores for two 16-key half tiles: S = Q(16x64) * K^T(64x16)
    float p[2][8];
    float corr[8];
#pragma unroll
    for (int t2 = 0; t2 < 2; ++t2) {
      const int z = zBase + t2 * 16 + nl;
      const _Float16* krow = base + (size_t)z * 3072 + D_ + h * HD_;
      const int kc = g * 16;
      v16h bf0 = *(const v16h*)(krow + kc);        // K rows are contiguous
      v16h bf1 = *(const v16h*)(krow + 32 + kc);   // => valid B-operand loads
      v8f c = {};
      c = wmma_f16(aq[0].v, bf0, c);
      c = wmma_f16(aq[1].v, bf1, c);
      const float lmk = lmb[z];
#pragma unroll
      for (int r = 0; r < 8; ++r) {
        const int m = rowBase + r + 8 * g;
        float s = c[r] * 0.125f + lmq[r] + lmk;    // 1/sqrt(64) = 0.125
        if (z > m) s = -INFINITY;                  // causal
        p[t2][r] = s;
      }
    }

    // Online softmax update (row stats replicated across the 16-lane half)
#pragma unroll
    for (int r = 0; r < 8; ++r) {
      float mx = fmaxf(p[0][r], p[1][r]);
#pragma unroll
      for (int off = 1; off < 16; off <<= 1)
        mx = fmaxf(mx, __shfl_xor(mx, off, 32));
      const float nm = fmaxf(rmax[r], mx);
      corr[r] = __expf(rmax[r] - nm);
      rmax[r] = nm;
      float p0 = __expf(p[0][r] - nm);
      float p1 = __expf(p[1][r] - nm);
      p[0][r] = p0; p[1][r] = p1;
      float ps = p0 + p1;
#pragma unroll
      for (int off = 1; off < 16; off <<= 1)
        ps += __shfl_xor(ps, off, 32);
      rsum[r] = rsum[r] * corr[r] + ps;
    }

    // Restage P (16x32) through LDS to build the next A operand
#pragma unroll
    for (int t2 = 0; t2 < 2; ++t2)
#pragma unroll
      for (int r = 0; r < 8; ++r)
        ldsP[(r + 8 * g) * 32 + t2 * 16 + nl] = (_Float16)p[t2][r];
    asm volatile("s_wait_dscnt 0" ::: "memory");   // cross-lane LDS hazard

    // Rescale running output, then O += P(16x32) * V(32x64)
#pragma unroll
    for (int dt = 0; dt < 4; ++dt)
#pragma unroll
      for (int r = 0; r < 8; ++r) o[dt][r] *= corr[r];

    AFrag ap;
    const int kb = g * 16;
    ap.h[0] = *(const v8h*)(ldsP + nl * 32 + kb);
    ap.h[1] = *(const v8h*)(ldsP + nl * 32 + kb + 8);
#pragma unroll
    for (int dt = 0; dt < 4; ++dt) {
      v16h bv = *(const v16h*)(ldsV + (dt * 16 + nl) * 32 + g * 16);
      o[dt] = wmma_f16(ap.v, bv, o[dt]);
    }
  }

  // Epilogue: normalize and store f16 in (b,s,h,hd) layout for the out-proj
#pragma unroll
  for (int dt = 0; dt < 4; ++dt) {
#pragma unroll
    for (int r = 0; r < 8; ++r) {
      const int m = rowBase + r + 8 * g;
      const int d = h * HD_ + dt * 16 + nl;
      attnh[((size_t)b * S_ + m) * D_ + d] = (_Float16)(o[dt][r] / rsum[r]);
    }
  }
}

// ---------------------------------------------------------------- launcher
extern "C" void kernel_launch(void* const* d_in, const int* in_sizes, int n_in,
                              void* d_out, int out_size, void* d_ws,
                              size_t ws_size, hipStream_t stream) {
  (void)in_sizes; (void)n_in; (void)out_size; (void)ws_size;
  const float* x     = (const float*)d_in[0];
  const float* mask  = (const float*)d_in[1];
  const float* W_qkv = (const float*)d_in[2];
  const float* b_qkv = (const float*)d_in[3];
  const float* W_o   = (const float*)d_in[4];
  const float* b_o   = (const float*)d_in[5];

  char* ws = (char*)d_ws;
  auto alloc = [&](size_t bytes) -> char* {
    char* p = ws;
    ws += (bytes + 255) & ~(size_t)255;
    return p;
  };
  _Float16* xh    = (_Float16*)alloc((size_t)M_ * D_ * 2);    // x in f16
  _Float16* Wqkvt = (_Float16*)alloc((size_t)N3D_ * D_ * 2);  // W_qkv^T f16
  _Float16* Wot   = (_Float16*)alloc((size_t)D_ * D_ * 2);    // W_o^T f16
  _Float16* qkvh  = (_Float16*)alloc((size_t)M_ * N3D_ * 2);  // qkv f16
  _Float16* attnh = (_Float16*)alloc((size_t)M_ * D_ * 2);    // attn out f16
  float*    logm  = (float*)alloc((size_t)M_ * 4);            // log(mask)

  const int T = 256;
  cvt_f16_kernel<<<(M_ * D_ + T - 1) / T, T, 0, stream>>>(x, xh, M_ * D_);
  transpose_f16_kernel<<<(N3D_ * D_ + T - 1) / T, T, 0, stream>>>(W_qkv, Wqkvt,
                                                                  D_, N3D_);
  transpose_f16_kernel<<<(D_ * D_ + T - 1) / T, T, 0, stream>>>(W_o, Wot, D_,
                                                                D_);
  log_kernel<<<(B_ * S_ + T - 1) / T, T, 0, stream>>>(mask, logm, B_ * S_);

  // QKV projection: 4096 x 3072 x 1024
  gemm_kernel<true><<<(M_ / 32) * (N3D_ / 64) / 8, 256, 0, stream>>>(
      xh, Wqkvt, b_qkv, qkvh, M_, N3D_, D_);

  // Attention: one wave per (b, h, 16-row block) = 4096 waves
  attn_kernel<<<(B_ * H_ * (S_ / 16)) / 8, 256, 0, stream>>>(qkvh, logm, attnh);

  // Output projection: 4096 x 1024 x 1024, f32 output
  gemm_kernel<false><<<(M_ / 32) * (D_ / 64) / 8, 256, 0, stream>>>(
      attnh, Wot, b_o, d_out, M_, D_, D_);
}
